// DecoupledRouterPool_70446053589396
// MI455X (gfx1250) — matrix operand order repaired and checked by reading, not compiled, not run on hardware
//
#include <hip/hip_runtime.h>
#include <hip/hip_bf16.h>

// ---------------------------------------------------------------------------
// Types for WMMA fragments (CDNA5 / gfx1250, wave32)
// ---------------------------------------------------------------------------
typedef __attribute__((ext_vector_type(16))) __bf16 v16bf;
typedef __attribute__((ext_vector_type(8)))  float  v8f;
typedef unsigned int uint4v __attribute__((ext_vector_type(4)));

union AF { v16bf v; uint4v q[2]; };

// fp32 -> bf16 with round-to-nearest-even
static __device__ __forceinline__ unsigned short f2bf(float f) {
  union { float f; unsigned int u; } cv; cv.f = f;
  unsigned int u = cv.u;
  unsigned int r = u + 0x7FFFu + ((u >> 16) & 1u);
  return (unsigned short)(r >> 16);
}

// ---------------------------------------------------------------------------
// Pre-pass: fp32 -> bf16 (row-major)
// ---------------------------------------------------------------------------
__global__ void cvt_bf16_kernel(const float* __restrict__ src,
                                unsigned short* __restrict__ dst, int n) {
  int i = blockIdx.x * blockDim.x + threadIdx.x;
  if (i < n) dst[i] = f2bf(src[i]);
}

// W[c][out][in] (fp32) -> Wt[c][in][out] (bf16, k-major for WMMA B operand)
__global__ void cvt_transpose_kernel(const float* __restrict__ W,
                                     unsigned short* __restrict__ Wt,
                                     int C, int OUT, int IN) {
  int tid = blockIdx.x * blockDim.x + threadIdx.x;
  int total = C * OUT * IN;
  if (tid >= total) return;
  int per = OUT * IN;
  int c = tid / per;
  int r = tid - c * per;
  int k = r / OUT;      // input-dim index
  int n = r - k * OUT;  // output-dim index
  Wt[tid] = f2bf(W[((size_t)c * OUT + n) * IN + k]);
}

// ---------------------------------------------------------------------------
// WMMA GEMM, M=32 per wave (two 16-row A tiles share every B fragment):
// out[32][Ntot] = A[32][K] (LDS bf16) * W[K][Ntot] (global bf16 k-major) + bias
// A-fragment per ISA 7.12.2 (16-bit A 16x32): lanes 0-15: M=lane, K chunks
// [0..7],[16..23]; lanes 16-31: M=lane-16, K chunks [8..15],[24..31].
// B-fragment: lane -> N = lane%16, K in the same chunks as A (strided loads).
// C/D: VGPR r, lane l -> M = r + 8*(l>=16), N = l%16.
// ---------------------------------------------------------------------------
__device__ __forceinline__ void gemm32(const unsigned short* __restrict__ aLds,
                                       int lda,
                                       const unsigned short* __restrict__ wg,
                                       int K, int Ntot,
                                       const float* __restrict__ bias,
                                       float* __restrict__ outLds, int ldo,
                                       int lane) {
  const int row  = lane & 15;
  const int ka   = (lane < 16) ? 0 : 8;
  const int rg   = (lane < 16) ? 0 : 8;
  const int coll = lane & 15;
  for (int ng = 0; ng < Ntot; ng += 64) {     // 4 N-tiles per pass
    v8f acc0[4] = {};
    v8f acc1[4] = {};
    for (int k0 = 0; k0 < K; k0 += 32) {
      AF ua0, ua1;
      const unsigned short* ap = aLds + row * lda + k0 + ka;
      ua0.q[0] = *(const uint4v*)(ap);
      ua0.q[1] = *(const uint4v*)(ap + 16);
      const unsigned short* ap1 = ap + 16 * lda;   // rows 16..31
      ua1.q[0] = *(const uint4v*)(ap1);
      ua1.q[1] = *(const uint4v*)(ap1 + 16);
      const unsigned short* bp = wg + (size_t)(k0 + ka) * Ntot + ng + coll;
      __builtin_prefetch(bp + (size_t)32 * Ntot, 0, 1);  // next K-step weights
#pragma unroll
      for (int t = 0; t < 4; ++t) {
        AF ub;
#pragma unroll
        for (int s = 0; s < 4; ++s) {
          ub.q[0][s] = (unsigned)bp[(2 * s) * Ntot + t * 16] | ((unsigned)bp[(2 * s + 1) * Ntot + t * 16] << 16);
          ub.q[1][s] = (unsigned)bp[(16 + 2 * s) * Ntot + t * 16] | ((unsigned)bp[(17 + 2 * s) * Ntot + t * 16] << 16);
        }
        acc0[t] = __builtin_amdgcn_wmma_f32_16x16x32_bf16(
            false, ua0.v, false, ub.v, (short)0, acc0[t], false, false);
        acc1[t] = __builtin_amdgcn_wmma_f32_16x16x32_bf16(
            false, ua1.v, false, ub.v, (short)0, acc1[t], false, false);
      }
    }
#pragma unroll
    for (int t = 0; t < 4; ++t) {
      int col = ng + t * 16 + coll;
      float b = bias[col];
#pragma unroll
      for (int r = 0; r < 8; ++r) {
        outLds[(size_t)(r + rg) * ldo + col]      = acc0[t][r] + b;
        outLds[(size_t)(16 + r + rg) * ldo + col] = acc1[t][r] + b;
      }
    }
  }
}

// Layer-3 GEMM (M=32) that never materializes t: accumulates per-lane partial
// sum(t^2) and sum(t*anchor) straight from the WMMA accumulators.
__device__ __forceinline__ void gemm32_metrics(
    const unsigned short* __restrict__ aLds, int lda,
    const unsigned short* __restrict__ wg, int K, int Ntot,
    const float* __restrict__ bias, const float* __restrict__ anch,
    float tsq0[8], float dot0[8], float tsq1[8], float dot1[8], int lane) {
  const int row  = lane & 15;
  const int ka   = (lane < 16) ? 0 : 8;
  const int coll = lane & 15;
  for (int ng = 0; ng < Ntot; ng += 64) {
    v8f acc0[4] = {};
    v8f acc1[4] = {};
    for (int k0 = 0; k0 < K; k0 += 32) {
      AF ua0, ua1;
      const unsigned short* ap = aLds + row * lda + k0 + ka;
      ua0.q[0] = *(const uint4v*)(ap);
      ua0.q[1] = *(const uint4v*)(ap + 16);
      const unsigned short* ap1 = ap + 16 * lda;
      ua1.q[0] = *(const uint4v*)(ap1);
      ua1.q[1] = *(const uint4v*)(ap1 + 16);
      const unsigned short* bp = wg + (size_t)(k0 + ka) * Ntot + ng + coll;
      __builtin_prefetch(bp + (size_t)32 * Ntot, 0, 1);
#pragma unroll
      for (int t = 0; t < 4; ++t) {
        AF ub;
#pragma unroll
        for (int s = 0; s < 4; ++s) {
          ub.q[0][s] = (unsigned)bp[(2 * s) * Ntot + t * 16] | ((unsigned)bp[(2 * s + 1) * Ntot + t * 16] << 16);
          ub.q[1][s] = (unsigned)bp[(16 + 2 * s) * Ntot + t * 16] | ((unsigned)bp[(17 + 2 * s) * Ntot + t * 16] << 16);
        }
        acc0[t] = __builtin_amdgcn_wmma_f32_16x16x32_bf16(
            false, ua0.v, false, ub.v, (short)0, acc0[t], false, false);
        acc1[t] = __builtin_amdgcn_wmma_f32_16x16x32_bf16(
            false, ua1.v, false, ub.v, (short)0, acc1[t], false, false);
      }
    }
#pragma unroll
    for (int t = 0; t < 4; ++t) {
      int col = ng + t * 16 + coll;
      float b  = bias[col];
      float av = anch[col];
#pragma unroll
      for (int r = 0; r < 8; ++r) {
        float v0 = acc0[t][r] + b;
        float v1 = acc1[t][r] + b;
        tsq0[r] += v0 * v0;
        dot0[r] += v0 * av;
        tsq1[r] += v1 * v1;
        dot1[r] += v1 * av;
      }
    }
  }
}

// LayerNorm (population var, eps 1e-5) + exact GELU:
// h[ROWS][N] fp32 -> act[ROWS][N] bf16
__device__ __forceinline__ void ln_gelu(const float* __restrict__ hbuf,
                                        int ROWS, int N,
                                        unsigned short* __restrict__ actbuf,
                                        const float* __restrict__ g,
                                        const float* __restrict__ be, int lane) {
  for (int m = 0; m < ROWS; ++m) {
    float s = 0.f, ss = 0.f;
    for (int j = lane; j < N; j += 32) {
      float v = hbuf[m * N + j];
      s += v; ss += v * v;
    }
#pragma unroll
    for (int off = 16; off > 0; off >>= 1) {
      s  += __shfl_xor(s,  off);
      ss += __shfl_xor(ss, off);
    }
    float inv_n = 1.f / (float)N;
    float mu = s * inv_n;
    float var = ss * inv_n - mu * mu;
    float rstd = rsqrtf(var + 1e-5f);
    for (int j = lane; j < N; j += 32) {
      float v = (hbuf[m * N + j] - mu) * rstd * g[j] + be[j];
      float ge = 0.5f * v * (1.f + erff(v * 0.70710678118654752f));
      actbuf[m * N + j] = f2bf(ge);
    }
  }
}

// ---------------------------------------------------------------------------
// Fused expert kernel: grid = (B/128, C); 4 waves/block, 32 tokens/wave.
// Per-wave LDS (64 KB): [0,32K) x-tile bf16 [32][512] (later act bf16 [32][256]),
// [32K,64K) h fp32 [32][256].  Block total 256 KB (WGP has 320 KB).
// ---------------------------------------------------------------------------
__global__ __launch_bounds__(128) void moe_router_kernel(
    const unsigned short* __restrict__ xb,
    const unsigned short* __restrict__ W1t, const float* __restrict__ b1,
    const float* __restrict__ g1, const float* __restrict__ be1,
    const unsigned short* __restrict__ W2t, const float* __restrict__ b2,
    const float* __restrict__ g2, const float* __restrict__ be2,
    const unsigned short* __restrict__ W3t, const float* __restrict__ b3,
    const float* __restrict__ anchors,
    float* __restrict__ dist_out, float* __restrict__ sim_out) {
  extern __shared__ char smem[];
  const int lane = threadIdx.x & 31;
  const int wave = threadIdx.x >> 5;
  const int c = blockIdx.y;
  const int token0 = blockIdx.x * 128 + wave * 32;

  char* base = smem + (size_t)wave * 65536;
  unsigned short* xbuf   = (unsigned short*)base;            // [32][512] bf16
  unsigned short* actbuf = (unsigned short*)base;            // [32][256] bf16 (aliases dead x)
  float*          hbuf   = (float*)(base + 32768);           // [32][256] fp32

  // Stage this wave's 32 token rows (contiguous 32 KB) into LDS.
  {
    const uint4v* src = (const uint4v*)(xb + (size_t)token0 * 512);
    uint4v* dst = (uint4v*)xbuf;
    for (int i = lane; i < 2048; i += 32) dst[i] = src[i];
  }
  __syncthreads();

  // Layer 1: [32x512] x [512x256]
  gemm32(xbuf, 512, W1t + (size_t)c * 512 * 256, 512, 256,
         b1 + (size_t)c * 256, hbuf, 256, lane);
  __syncthreads();
  ln_gelu(hbuf, 32, 256, actbuf, g1 + (size_t)c * 256, be1 + (size_t)c * 256, lane);
  __syncthreads();

  // Layer 2: [32x256] x [256x256]
  gemm32(actbuf, 256, W2t + (size_t)c * 256 * 256, 256, 256,
         b2 + (size_t)c * 256, hbuf, 256, lane);
  __syncthreads();
  ln_gelu(hbuf, 32, 256, actbuf, g2 + (size_t)c * 256, be2 + (size_t)c * 256, lane);
  __syncthreads();

  // Layer 3: [32x256] x [256x512], metrics straight from accumulators.
  const float* a = anchors + (size_t)c * 512;
  float tsq0[8] = {}, dot0[8] = {}, tsq1[8] = {}, dot1[8] = {};
  gemm32_metrics(actbuf, 256, W3t + (size_t)c * 256 * 512, 256, 512,
                 b3 + (size_t)c * 512, a, tsq0, dot0, tsq1, dot1, lane);

  // Anchor norm (full-wave reduction, same value for all lanes).
  float asq = 0.f;
  for (int j = lane; j < 512; j += 32) { float av = a[j]; asq += av * av; }
#pragma unroll
  for (int off = 16; off > 0; off >>= 1) asq += __shfl_xor(asq, off);
  float rasq = rsqrtf(asq);

  // Reduce per-row partials across each 16-lane half (disjoint column sets).
  // Lanes 0-15 hold rows {mt*16 + 0..7}; lanes 16-31 hold rows {mt*16 + 8..15}.
#pragma unroll
  for (int r = 0; r < 8; ++r) {
#pragma unroll
    for (int off = 8; off > 0; off >>= 1) {
      tsq0[r] += __shfl_xor(tsq0[r], off);
      dot0[r] += __shfl_xor(dot0[r], off);
      tsq1[r] += __shfl_xor(tsq1[r], off);
      dot1[r] += __shfl_xor(dot1[r], off);
    }
  }
  // sim = t.a/(|t||a|); dist = |t/|t| - a/|a|| = sqrt(2 - 2*sim).
  if ((lane & 15) == 0) {
    const int mrow = (lane < 16) ? 0 : 8;   // row offset within each 16-tile
#pragma unroll
    for (int r = 0; r < 8; ++r) {
      {
        float sim = dot0[r] * rsqrtf(tsq0[r]) * rasq;
        float d = sqrtf(fmaxf(2.f - 2.f * sim, 0.f));
        size_t tkn = (size_t)(token0 + mrow + r);
        dist_out[tkn * 100 + c] = d;
        sim_out[tkn * 100 + c]  = sim;
      }
      {
        float sim = dot1[r] * rsqrtf(tsq1[r]) * rasq;
        float d = sqrtf(fmaxf(2.f - 2.f * sim, 0.f));
        size_t tkn = (size_t)(token0 + 16 + mrow + r);
        dist_out[tkn * 100 + c] = d;
        sim_out[tkn * 100 + c]  = sim;
      }
    }
  }
}

// routed[t] = argmin_c dist[t][c] (first index on ties, matching argmin)
__global__ void argmin_kernel(const float* __restrict__ dist,
                              float* __restrict__ routed, int B, int C) {
  int t = blockIdx.x * blockDim.x + threadIdx.x;
  if (t >= B) return;
  float best = dist[(size_t)t * C];
  int bi = 0;
  for (int c = 1; c < C; ++c) {
    float d = dist[(size_t)t * C + c];
    if (d < best) { best = d; bi = c; }
  }
  routed[t] = (float)bi;
}

// ---------------------------------------------------------------------------
// Host launcher
// ---------------------------------------------------------------------------
extern "C" void kernel_launch(void* const* d_in, const int* in_sizes, int n_in,
                              void* d_out, int out_size, void* d_ws, size_t ws_size,
                              hipStream_t stream) {
  (void)in_sizes; (void)n_in; (void)out_size; (void)ws_size;
  const float* x       = (const float*)d_in[0];
  const float* W1      = (const float*)d_in[1];
  const float* b1      = (const float*)d_in[2];
  const float* g1      = (const float*)d_in[3];
  const float* be1     = (const float*)d_in[4];
  const float* W2      = (const float*)d_in[5];
  const float* b2      = (const float*)d_in[6];
  const float* g2      = (const float*)d_in[7];
  const float* be2     = (const float*)d_in[8];
  const float* W3      = (const float*)d_in[9];
  const float* b3      = (const float*)d_in[10];
  const float* anchors = (const float*)d_in[11];

  // Workspace: bf16 x + k-major bf16 weights (~64.5 MB)
  unsigned short* xb  = (unsigned short*)d_ws;
  unsigned short* W1t = xb  + (size_t)2048 * 512;
  unsigned short* W2t = W1t + (size_t)100 * 512 * 256;
  unsigned short* W3t = W2t + (size_t)100 * 256 * 256;

  float* out    = (float*)d_out;
  float* routed = out;                       // [2048]
  float* dist   = out + 2048;                // [2048][100]
  float* sims   = dist + (size_t)2048 * 100; // [2048][100]

  // Pre-pass conversions
  cvt_bf16_kernel<<<(2048 * 512) / 256, 256, 0, stream>>>(x, xb, 2048 * 512);
  cvt_transpose_kernel<<<(100 * 256 * 512) / 256, 256, 0, stream>>>(W1, W1t, 100, 256, 512);
  cvt_transpose_kernel<<<(100 * 256 * 256) / 256, 256, 0, stream>>>(W2, W2t, 100, 256, 256);
  cvt_transpose_kernel<<<(100 * 512 * 256) / 256, 256, 0, stream>>>(W3, W3t, 100, 512, 256);

  // Fused expert pipeline: 256 KB dynamic LDS per block (WGP has 320 KB)
  const int shmem = 4 * 65536;
  (void)hipFuncSetAttribute((const void*)moe_router_kernel,
                            hipFuncAttributeMaxDynamicSharedMemorySize, shmem);
  moe_router_kernel<<<dim3(2048 / 128, 100), 128, shmem, stream>>>(
      xb, W1t, b1, g1, be1, W2t, b2, g2, be2, W3t, b3, anchors, dist, sims);

  argmin_kernel<<<(2048 + 255) / 256, 256, 0, stream>>>(dist, routed, 2048, 100);
}
